// SelectiveSSM_21758304322225
// MI455X (gfx1250) — matrix operand order repaired and checked by reading, not compiled, not run on hardware
//
#include <hip/hip_runtime.h>
#include <cstdint>

// Problem constants (match reference)
#define D_INNER  2048
#define D_STATE  16
#define DT_RANK  128
#define BATCH    2
#define SEQ_LEN  2048
#define E_DIM    (DT_RANK + 2 * D_STATE)   // 160
#define NCHUNK   32
#define CHUNK    64                        // SEQ_LEN / NCHUNK

typedef __attribute__((ext_vector_type(16))) _Float16 v16h;
typedef __attribute__((ext_vector_type(8)))  float    v8f;

typedef __attribute__((ext_vector_type(4))) unsigned int tdm_u32x4;
typedef __attribute__((ext_vector_type(8))) int          tdm_i32x8;
typedef __attribute__((ext_vector_type(4))) int          tdm_i32x4;

// ---- Tensor Data Mover: DMA a 2D f32 tile (tile_w x tile_h, row stride in
// elements) from global memory into LDS at lds_off_bytes. Issued per-wave;
// completion tracked by TENSORcnt. D# packing per CDNA5 ISA §8 (08_async_tensor).
// This toolchain exposes the 6-arg builtin: (g0,u32x4)(g1,i32x8)(i32x4)(i32x4)(i32x8)(cpol).
__device__ __forceinline__ void tdm_load_tile_2d_f32(unsigned lds_off_bytes,
                                                     const void* gaddr,
                                                     unsigned tile_w_elems,
                                                     unsigned tile_h_rows,
                                                     unsigned row_stride_elems) {
    unsigned long long ga = (unsigned long long)(uintptr_t)gaddr;
    tdm_u32x4 g0;
    g0[0] = 1u;                                    // count=1 (valid), no gather
    g0[1] = lds_off_bytes;                         // lds_addr (bytes)
    g0[2] = (unsigned)(ga & 0xFFFFFFFFu);          // global_addr[31:0]
    g0[3] = (unsigned)((ga >> 32) & 0x1FFFFFFu)    // global_addr[56:32]
            | (2u << 30);                          // type=2 ("image")
    tdm_i32x8 g1;
    g1[0] = (2 << 16);                             // data_size=4B; no mcast/pad/iter
    g1[1] = (int)((tile_w_elems & 0xFFFFu) << 16); // tensor_dim0[15:0] (== tile_w)
    g1[2] = (int)((tile_h_rows & 0xFFFFu) << 16);  // tensor_dim0 hi=0 | tensor_dim1 lo
    g1[3] = (int)((tile_w_elems & 0xFFFFu) << 16); // tensor_dim1 hi=0 | tile_dim0
    g1[4] = (int)(tile_h_rows & 0xFFFFu);          // tile_dim1 | tile_dim2=0
    g1[5] = (int)row_stride_elems;                 // tensor_dim0_stride[31:0]
    g1[6] = 0;                                     // stride0 hi | stride1 lo
    g1[7] = 0;                                     // stride1 hi
    tdm_i32x4 gz4 = {0, 0, 0, 0};                  // groups 2/3 unused (2D tensor)
    tdm_i32x8 gz8 = {0, 0, 0, 0, 0, 0, 0, 0};
    __builtin_amdgcn_tensor_load_to_lds(g0, g1, gz4, gz4, gz8, 0);
}

// ---- WMMA fragment loaders (f32 memory -> f16 fragments, f32 accumulate) ----
// A fragment (contiguous k order used here, not ISA table order): lane m: elems 0..15 = 16 consecutive K at
// koff = (lane>>4)*16. p points at &A[row][k0+koff].
__device__ __forceinline__ v16h frag_a_from_f32(const float* __restrict__ p) {
    const float4* q0 = (const float4*)p;
    const float4* q1 = (const float4*)(p + 8);
    float4 a = q0[0], b = q0[1], c = q1[0], d = q1[1];
    v16h v;
    v[0]=(_Float16)a.x;  v[1]=(_Float16)a.y;  v[2]=(_Float16)a.z;  v[3]=(_Float16)a.w;
    v[4]=(_Float16)b.x;  v[5]=(_Float16)b.y;  v[6]=(_Float16)b.z;  v[7]=(_Float16)b.w;
    v[8]=(_Float16)c.x;  v[9]=(_Float16)c.y;  v[10]=(_Float16)c.z; v[11]=(_Float16)c.w;
    v[12]=(_Float16)d.x; v[13]=(_Float16)d.y; v[14]=(_Float16)d.z; v[15]=(_Float16)d.w;
    return v;
}
// B fragment (same contiguous k order as A): lane n=lane&15, elems 0..15 = 16 consecutive K at
// koff = (lane>>4)*16. B[k][n] = W[n][k]. p points at &W[n][k0+koff].
__device__ __forceinline__ v16h frag_b_from_f32(const float* __restrict__ p) {
    const float4* q = (const float4*)p;
    float4 a = q[0], b = q[1], c = q[2], d = q[3];
    v16h v;
    v[0]=(_Float16)a.x;  v[1]=(_Float16)a.y;  v[2]=(_Float16)a.z;  v[3]=(_Float16)a.w;
    v[4]=(_Float16)b.x;  v[5]=(_Float16)b.y;  v[6]=(_Float16)b.z;  v[7]=(_Float16)b.w;
    v[8]=(_Float16)c.x;  v[9]=(_Float16)c.y;  v[10]=(_Float16)c.z; v[11]=(_Float16)c.w;
    v[12]=(_Float16)d.x; v[13]=(_Float16)d.y; v[14]=(_Float16)d.z; v[15]=(_Float16)d.w;
    return v;
}

// ---- Kernel A: x_dbl[M=4096][160] = x[4096][2048] @ x_proj_w[160][2048]^T ----
__global__ void __launch_bounds__(128)
ssm_proj1_wmma(const float* __restrict__ x, const float* __restrict__ w,
               float* __restrict__ xdbl) {
    const int lane = threadIdx.x;                                // 0..31
    const int mt   = blockIdx.x * blockDim.y + threadIdx.y;      // 0..255
    const int nt   = blockIdx.y;                                 // 0..9
    const int m0 = mt * 16, n0 = nt * 16;
    const int row  = m0 + (lane & 15);
    const int ncol = n0 + (lane & 15);
    const float* ap = x + (size_t)row * D_INNER + ((lane >> 4) * 16);
    const float* bp = w + (size_t)ncol * D_INNER + ((lane >> 4) * 16);
    v8f acc = {};
    for (int k0 = 0; k0 < D_INNER; k0 += 32) {
        v16h af = frag_a_from_f32(ap + k0);
        v16h bf = frag_b_from_f32(bp + k0);
        acc = __builtin_amdgcn_wmma_f32_16x16x32_f16(false, af, false, bf,
                                                     (short)0, acc, false, false);
    }
    // C/D layout: VGPR r -> M = m0 + r + 8*(lane>>4), N = n0 + (lane&15)
    const int mbase = m0 + ((lane >> 4) << 3);
    #pragma unroll
    for (int r = 0; r < 8; ++r)
        xdbl[(size_t)(mbase + r) * E_DIM + ncol] = acc[r];
}

// ---- Kernel B: dt[4096][2048] = softplus(x_dbl[:, :128] @ dt_proj_w^T + b) ----
__global__ void __launch_bounds__(128)
ssm_proj2_wmma(const float* __restrict__ xdbl, const float* __restrict__ w2,
               const float* __restrict__ bias, float* __restrict__ dtws) {
    const int lane = threadIdx.x;
    const int mt   = blockIdx.x * blockDim.y + threadIdx.y;      // 0..255
    const int nt   = blockIdx.y;                                 // 0..127
    const int m0 = mt * 16, n0 = nt * 16;
    const int row  = m0 + (lane & 15);
    const int ncol = n0 + (lane & 15);
    const float* ap = xdbl + (size_t)row * E_DIM + ((lane >> 4) * 16);
    const float* bp = w2 + (size_t)ncol * DT_RANK + ((lane >> 4) * 16);
    v8f acc = {};
    #pragma unroll
    for (int k0 = 0; k0 < DT_RANK; k0 += 32) {
        v16h af = frag_a_from_f32(ap + k0);
        v16h bf = frag_b_from_f32(bp + k0);
        acc = __builtin_amdgcn_wmma_f32_16x16x32_f16(false, af, false, bf,
                                                     (short)0, acc, false, false);
    }
    const float bz = bias[ncol];
    const int mbase = m0 + ((lane >> 4) << 3);
    #pragma unroll
    for (int r = 0; r < 8; ++r) {
        float z  = acc[r] + bz;
        float sp = (z > 20.0f) ? z : log1pf(__expf(z));          // softplus
        dtws[(size_t)(mbase + r) * D_INNER + ncol] = sp;
    }
}

// ---- Kernel C: per-chunk scan summaries (h0 = 0): (prod dA, h_end) ----
// One 256-thread block per (b, chunk, d-block of 256). The block's B vectors
// (64 steps x 32 floats, stride 160 in x_dbl) are DMA'd to LDS by the TDM.
__global__ void __launch_bounds__(256)
ssm_scan_pass1(const float* __restrict__ x, const float* __restrict__ xdbl,
               const float* __restrict__ dtws, const float* __restrict__ A_log,
               float* __restrict__ summ) {
    const int blk   = blockIdx.x;                  // 0..511
    const int rest  = blk >> 3;                    // uniform: (b, chunk)
    const int chunk = rest & (NCHUNK - 1);
    const int b     = rest >> 5;
    const int d     = ((blk & 7) << 8) + threadIdx.x;

    const size_t rowbase = (size_t)b * SEQ_LEN + (size_t)chunk * CHUNK;

    __shared__ __align__(16) float bc_lds[CHUNK * 32];           // 8 KB tile
    if (threadIdx.x < 32) {                        // wave 0 issues the TDM DMA
        tdm_load_tile_2d_f32((unsigned)(uintptr_t)&bc_lds[0],
                             xdbl + rowbase * E_DIM + DT_RANK,
                             32, CHUNK, E_DIM);
        __builtin_amdgcn_s_wait_tensorcnt(0);
    }
    __syncthreads();

    float Ar[D_STATE];
    #pragma unroll
    for (int n = 0; n < D_STATE; ++n)
        Ar[n] = -__expf(A_log[d * D_STATE + n]);                 // A = -exp(A_log)

    float h[D_STATE], apd[D_STATE];
    #pragma unroll
    for (int n = 0; n < D_STATE; ++n) { h[n] = 0.0f; apd[n] = 1.0f; }

    const float* dtp = dtws + rowbase * D_INNER + d;
    const float* xp  = x    + rowbase * D_INNER + d;

    for (int s = 0; s < CHUNK; ++s) {
        if (s + 8 < CHUNK) {  // stride-8KB streams: software prefetch (global_prefetch_b8)
            __builtin_prefetch(dtp + (size_t)(s + 8) * D_INNER, 0, 0);
            __builtin_prefetch(xp  + (size_t)(s + 8) * D_INNER, 0, 0);
        }
        const float dt = dtp[(size_t)s * D_INNER];
        const float xv = xp[(size_t)s * D_INNER];
        const float4* Bq = (const float4*)&bc_lds[s * 32];       // LDS broadcast
        float4 B0 = Bq[0], B1 = Bq[1], B2 = Bq[2], B3 = Bq[3];
        const float Bv[D_STATE] = {B0.x,B0.y,B0.z,B0.w, B1.x,B1.y,B1.z,B1.w,
                                   B2.x,B2.y,B2.z,B2.w, B3.x,B3.y,B3.z,B3.w};
        const float dx = dt * xv;
        #pragma unroll
        for (int n = 0; n < D_STATE; ++n) {
            float da = __expf(dt * Ar[n]);
            h[n]   = fmaf(da, h[n], dx * Bv[n]);
            apd[n] *= da;
        }
    }
    const size_t sb = ((((size_t)b * D_INNER + d) * NCHUNK) + chunk) * (2 * D_STATE);
    #pragma unroll
    for (int n = 0; n < D_STATE; ++n) {
        summ[sb + 2 * n + 0] = apd[n];
        summ[sb + 2 * n + 1] = h[n];
    }
}

// ---- Kernel D: sequential combine over 32 chunks; writes entry states in place ----
__global__ void __launch_bounds__(256)
ssm_scan_pass2(float* __restrict__ summ) {
    const int idx = blockIdx.x * 256 + threadIdx.x;              // 0..65535
    const int n   = idx & (D_STATE - 1);
    const int d   = (idx >> 4) & (D_INNER - 1);
    const int b   = idx >> 15;
    size_t base = (((size_t)b * D_INNER + d) * NCHUNK) * (2 * D_STATE) + 2 * n;
    float h = 0.0f;
    #pragma unroll 4
    for (int c = 0; c < NCHUNK; ++c) {
        float a  = summ[base + 0];
        float he = summ[base + 1];
        summ[base + 0] = h;          // entry state for chunk c (overwrites A-prod)
        h = fmaf(a, h, he);
        base += 2 * D_STATE;
    }
}

// ---- Kernel E: re-scan each chunk from its entry state, emit y + D*x ----
__global__ void __launch_bounds__(256)
ssm_scan_pass3(const float* __restrict__ x, const float* __restrict__ xdbl,
               const float* __restrict__ dtws, const float* __restrict__ A_log,
               const float* __restrict__ Dvec, const float* __restrict__ summ,
               float* __restrict__ y) {
    const int blk   = blockIdx.x;
    const int rest  = blk >> 3;
    const int chunk = rest & (NCHUNK - 1);
    const int b     = rest >> 5;
    const int d     = ((blk & 7) << 8) + threadIdx.x;

    const size_t rowbase = (size_t)b * SEQ_LEN + (size_t)chunk * CHUNK;

    __shared__ __align__(16) float bc_lds[CHUNK * 32];           // B|C tile, 8 KB
    if (threadIdx.x < 32) {
        tdm_load_tile_2d_f32((unsigned)(uintptr_t)&bc_lds[0],
                             xdbl + rowbase * E_DIM + DT_RANK,
                             32, CHUNK, E_DIM);
        __builtin_amdgcn_s_wait_tensorcnt(0);
    }
    __syncthreads();

    float Ar[D_STATE], h[D_STATE];
    #pragma unroll
    for (int n = 0; n < D_STATE; ++n)
        Ar[n] = -__expf(A_log[d * D_STATE + n]);
    const size_t sb = ((((size_t)b * D_INNER + d) * NCHUNK) + chunk) * (2 * D_STATE);
    #pragma unroll
    for (int n = 0; n < D_STATE; ++n)
        h[n] = summ[sb + 2 * n];                                  // entry state

    const float* dtp = dtws + rowbase * D_INNER + d;
    const float* xp  = x    + rowbase * D_INNER + d;
    float*       yp  = y    + rowbase * D_INNER + d;
    const float Dd = Dvec[d];

    for (int s = 0; s < CHUNK; ++s) {
        if (s + 8 < CHUNK) {
            __builtin_prefetch(dtp + (size_t)(s + 8) * D_INNER, 0, 0);
            __builtin_prefetch(xp  + (size_t)(s + 8) * D_INNER, 0, 0);
        }
        const float dt = dtp[(size_t)s * D_INNER];
        const float xv = xp[(size_t)s * D_INNER];
        const float4* Bq = (const float4*)&bc_lds[s * 32];        // B = [0..15]
        const float4* Cq = (const float4*)&bc_lds[s * 32 + 16];   // C = [16..31]
        float4 B0 = Bq[0], B1 = Bq[1], B2 = Bq[2], B3 = Bq[3];
        float4 C0 = Cq[0], C1 = Cq[1], C2 = Cq[2], C3 = Cq[3];
        const float Bv[D_STATE] = {B0.x,B0.y,B0.z,B0.w, B1.x,B1.y,B1.z,B1.w,
                                   B2.x,B2.y,B2.z,B2.w, B3.x,B3.y,B3.z,B3.w};
        const float Cv[D_STATE] = {C0.x,C0.y,C0.z,C0.w, C1.x,C1.y,C1.z,C1.w,
                                   C2.x,C2.y,C2.z,C2.w, C3.x,C3.y,C3.z,C3.w};
        const float dx = dt * xv;
        float acc = 0.0f;
        #pragma unroll
        for (int n = 0; n < D_STATE; ++n) {
            float da = __expf(dt * Ar[n]);
            h[n] = fmaf(da, h[n], dx * Bv[n]);
            acc  = fmaf(h[n], Cv[n], acc);
        }
        yp[(size_t)s * D_INNER] = fmaf(Dd, xv, acc);
    }
}

extern "C" void kernel_launch(void* const* d_in, const int* in_sizes, int n_in,
                              void* d_out, int out_size, void* d_ws, size_t ws_size,
                              hipStream_t stream) {
    const float* x    = (const float*)d_in[0];   // (2,2048,2048)
    const float* xpw  = (const float*)d_in[1];   // (160,2048)
    const float* dpw  = (const float*)d_in[2];   // (2048,128)
    const float* dpb  = (const float*)d_in[3];   // (2048,)
    const float* alog = (const float*)d_in[4];   // (2048,16)
    const float* Dv   = (const float*)d_in[5];   // (2048,)
    float* y = (float*)d_out;

    // Workspace layout (fully overwritten each call): ~53 MB
    float* xdbl = (float*)d_ws;                                       // 4096*160
    float* dtws = xdbl + (size_t)BATCH * SEQ_LEN * E_DIM;             // 4096*2048
    float* summ = dtws + (size_t)BATCH * SEQ_LEN * D_INNER;           // 2*2048*32*32

    const dim3 blk(32, 4);  // 4 independent waves per workgroup, one 16x16 tile each
    ssm_proj1_wmma<<<dim3((BATCH * SEQ_LEN / 16) / 4, E_DIM / 16), blk, 0, stream>>>(
        x, xpw, xdbl);
    ssm_proj2_wmma<<<dim3((BATCH * SEQ_LEN / 16) / 4, D_INNER / 16), blk, 0, stream>>>(
        xdbl, dpw, dpb, dtws);
    ssm_scan_pass1<<<(BATCH * NCHUNK * D_INNER) / 256, 256, 0, stream>>>(
        x, xdbl, dtws, alog, summ);
    ssm_scan_pass2<<<(BATCH * D_INNER * D_STATE) / 256, 256, 0, stream>>>(summ);
    ssm_scan_pass3<<<(BATCH * NCHUNK * D_INNER) / 256, 256, 0, stream>>>(
        x, xdbl, dtws, alog, Dv, summ, y);
}
